// LSTM_19902878449909
// MI455X (gfx1250) — matrix operand order repaired and checked by reading, not compiled, not run on hardware
//
#include <hip/hip_runtime.h>
#include <hip/hip_bf16.h>
#include <cstdint>

// ---------------------------------------------------------------------------
// LSTM on MI455X (gfx1250), wave32 + WMMA bf16.
// Phase 0: one-time packs: x -> bf16 [T*B,D] (GEMM row order),
//          gate weights -> bf16 TRANSPOSED [4H,D]/[4H,H], bias -> [4H] fp32.
// Phase 1: x_proj = x @ Wi + bias. Double-buffered LDS tiles, one barrier
//          per K-step, B-fragments batched 4-at-a-time before WMMA issue.
// Phase 2: persistent recurrence, Wh slice LDS-resident, c-state register-
//          resident, double-buffered h stage, one grid barrier per step.
// All WMMA fragments load as 2x ds_load_b128.
// Workspace ~1.35 GB (bf16 x copy 256 MB + bf16 x_proj 1 GB + weights).
// ---------------------------------------------------------------------------

typedef __bf16 bf16_t;
typedef __attribute__((ext_vector_type(2)))  __bf16 v2bf;
typedef __attribute__((ext_vector_type(16))) __bf16 v16bf;
typedef __attribute__((ext_vector_type(8)))  float  v8f;

union FragU { uint4 q[2]; v16bf v; };   // 32 bytes

#define B_   128
#define T_   1024
#define D_   1024
#define H_   1024
#define G4H  4096
#define NWG  64
#define WHS_STRIDE 1032                 // 1024 + 8 pad (bank spread)

__device__ __forceinline__ float sigmoidf_(float x) {
  return 1.0f / (1.0f + __expf(-x));
}
// branch-free tanh: clamp then (e^2x - 1)/(e^2x + 1)
__device__ __forceinline__ float tanhf_(float x) {
  float xc = __builtin_fminf(15.0f, __builtin_fmaxf(-15.0f, x));
  float t  = __expf(2.0f * xc);
  return (t - 1.0f) / (t + 1.0f);
}
__device__ __forceinline__ unsigned pk2(float a, float b) {
  v2bf p;
  p.x = (bf16_t)a;
  p.y = (bf16_t)b;
  return __builtin_bit_cast(unsigned, p);
}

// ---- pack x: fp32 [B,T,D] -> bf16 [T*B, D] (row r = t*B + b) --------------
__global__ void __launch_bounds__(256) pack_x(
    const float* __restrict__ x, bf16_t* __restrict__ xb) {
  size_t u = (size_t)blockIdx.x * 256 + threadIdx.x;  // uint4 index
  size_t e = u * 8;                                   // element index
  int d = (int)(e & 1023);
  int r = (int)(e >> 10);
  int t = r >> 7;            // / B_
  int b = r & 127;           // % B_
  const float4* s = (const float4*)(x + (((size_t)b << 10) + (size_t)t) * 1024 + d);
  float4 f0 = s[0], f1 = s[1];
  uint4 p;
  p.x = pk2(f0.x, f0.y); p.y = pk2(f0.z, f0.w);
  p.z = pk2(f1.x, f1.y); p.w = pk2(f1.z, f1.w);
  *(uint4*)(xb + e) = p;
}

// ---- pack four fp32 [1024,1024] gate weights -> bf16 TRANSPOSED [4096,1024]
__global__ void __launch_bounds__(256) pack_w4t(
    const float* __restrict__ w0, const float* __restrict__ w1,
    const float* __restrict__ w2, const float* __restrict__ w3,
    bf16_t* __restrict__ dst) {
  unsigned i = blockIdx.x * 256u + threadIdx.x;   // 0 .. 4M-1
  unsigned k = i & 1023u;
  unsigned g = i >> 10;
  unsigned c = g & 1023u;
  unsigned gate = g >> 10;
  const float* s = (gate == 0) ? w0 : (gate == 1) ? w1 : (gate == 2) ? w2 : w3;
  dst[(size_t)g * 1024u + k] = (bf16_t)s[(size_t)k * 1024u + c];
}

// ---- pack bias: 4 x [1024] -> [4096] fp32 ---------------------------------
__global__ void __launch_bounds__(256) pack_bias(
    const float* __restrict__ bi, const float* __restrict__ bf,
    const float* __restrict__ bg, const float* __restrict__ bo,
    float* __restrict__ dst) {
  int i = blockIdx.x * 256 + threadIdx.x;   // 0..4095
  int c = i & 1023;
  int g = i >> 10;
  const float* s = (g == 0) ? bi : (g == 1) ? bf : (g == 2) ? bg : bo;
  dst[i] = s[c];
}

// ---- Phase 1: x_proj[r, g] = sum_d xb[r,d] * Wi[d,g] + bias[g] ------------
// 128x128 tile / block, 8 waves, each wave: 16 rows x 128 cols.
__global__ void __launch_bounds__(256) xproj_gemm(
    const bf16_t* __restrict__ xb,     // [T*B, D] bf16
    const bf16_t* __restrict__ Wibt,   // [4H, D] bf16 (transposed)
    const float*  __restrict__ bias4,  // [4H] fp32
    bf16_t* __restrict__ xproj) {      // [T*B, 4H] bf16
  __shared__ bf16_t As[2][128][32];    // [buf][row][k]
  __shared__ bf16_t Bs[2][128][40];    // [buf][col][k], padded

  const int tid  = threadIdx.x;
  const int lane = tid & 31;
  const int wave = tid >> 5;
  const int ln   = lane & 15;
  const int hi   = lane >> 4;
  const int n0   = blockIdx.x * 128;
  const int m0   = blockIdx.y * 128;

  v8f acc[8] = {};

  // staging: 128 rows x 32 k per matrix; 2 threads per row/col
  const int ar0 = tid >> 1;
  const int ah  = (tid & 1) * 16;

  auto stage = [&](int buf, int kb) {
    const uint4* a4 = (const uint4*)(xb + (size_t)(m0 + ar0) * D_ + kb + ah);
    uint4 a0 = a4[0], a1 = a4[1];
    *(uint4*)&As[buf][ar0][ah]     = a0;
    *(uint4*)&As[buf][ar0][ah + 8] = a1;
    const uint4* b4 = (const uint4*)(Wibt + (size_t)(n0 + ar0) * D_ + kb + ah);
    uint4 u0 = b4[0], u1 = b4[1];
    *(uint4*)&Bs[buf][ar0][ah]     = u0;
    *(uint4*)&Bs[buf][ar0][ah + 8] = u1;
  };

  stage(0, 0);
  const int ar = wave * 16 + ln;
  for (int ki = 0; ki < D_ / 32; ++ki) {
    __syncthreads();
    if (ki + 1 < D_ / 32) stage((ki + 1) & 1, (ki + 1) * 32);
    const int buf = ki & 1;

    FragU fa;   // lane ln row, K = {hi*8..+7, 16+hi*8..+7}
    fa.q[0] = *(const uint4*)&As[buf][ar][hi * 8];
    fa.q[1] = *(const uint4*)&As[buf][ar][16 + hi * 8];
#pragma unroll
    for (int g = 0; g < 2; ++g) {
      FragU fb[4];
#pragma unroll
      for (int c = 0; c < 4; ++c) {
        const int cc = g * 4 + c;
        fb[c].q[0] = *(const uint4*)&Bs[buf][cc * 16 + ln][hi * 8];
        fb[c].q[1] = *(const uint4*)&Bs[buf][cc * 16 + ln][16 + hi * 8];
      }
#pragma unroll
      for (int c = 0; c < 4; ++c) {
        acc[g * 4 + c] = __builtin_amdgcn_wmma_f32_16x16x32_bf16(
            false, fa.v, false, fb[c].v, (short)0, acc[g * 4 + c], false, false);
      }
    }
  }

  // epilogue: + bias, store bf16. C/D layout: M = hi*8 + r, N = ln.
#pragma unroll
  for (int c = 0; c < 8; ++c) {
    const int colg = n0 + c * 16 + ln;
    const float bias = bias4[colg];
#pragma unroll
    for (int r = 0; r < 8; ++r) {
      const int row = m0 + wave * 16 + hi * 8 + r;
      xproj[(size_t)row * G4H + colg] = (bf16_t)(acc[c][r] + bias);
    }
  }
}

// ---- device-scope grid barrier (monotonic counter, reset per launch) ------
__device__ __forceinline__ void grid_barrier(unsigned* cnt, unsigned target) {
  __syncthreads();
  if (threadIdx.x == 0) {
    __threadfence();
    __hip_atomic_fetch_add(cnt, 1u, __ATOMIC_ACQ_REL, __HIP_MEMORY_SCOPE_AGENT);
    while (__hip_atomic_load(cnt, __ATOMIC_ACQUIRE, __HIP_MEMORY_SCOPE_AGENT) < target) {
      __builtin_amdgcn_s_sleep(2);
    }
  }
  __syncthreads();
}

// ---- Phase 2: persistent recurrence --------------------------------------
// WG wg owns h columns [wg*16, wg*16+16). LDS: Wh slice column-major
// (64 g-cols x 1024 k, padded) ~129 KB resident for all 1024 steps +
// 2 x 10 KB double-buffered h stage. Wave w owns batch rows [w*16,w*16+16);
// its 16x16 c-state tile lives in a v8f accumulator for the whole kernel.
__global__ void __launch_bounds__(256) lstm_recur(
    const bf16_t* __restrict__ Whbt,   // [4H,H] bf16 (transposed)
    const bf16_t* __restrict__ xproj,  // [T*B,4H] bf16
    bf16_t* __restrict__ h0, bf16_t* __restrict__ h1,  // [B,H] double buffer
    unsigned* __restrict__ cnt,
    float* __restrict__ out_h, float* __restrict__ out_c) {
  extern __shared__ char smem[];
  bf16_t* WhS = (bf16_t*)smem;         // [64][WHS_STRIDE] col-major slice
  bf16_t (*hS)[128][40] =
      (bf16_t(*)[128][40])(smem + (size_t)64 * WHS_STRIDE * sizeof(bf16_t));

  const int tid  = threadIdx.x;
  const int lane = tid & 31;
  const int wave = tid >> 5;
  const int ln   = lane & 15;
  const int hi   = lane >> 4;
  const int wg   = blockIdx.x;
  const int hc0  = wg * 16;
  const int rw0  = wave * 16;
  const int rowg0 = rw0 + hi * 8;      // + r -> global batch row

  // resident Wh slice -> LDS (once): 64 rows (g-cols) x 1024 k, b128 copies
  for (int u = tid; u < 64 * (H_ / 8); u += 256) {   // 8192 uint4
    int row = u >> 7;                  // 0..63  (= q*16 + c)
    int seg = u & 127;
    int q = row >> 4;
    int c = row & 15;
    int gcol = q * 1024 + hc0 + c;
    uint4 w = ((const uint4*)(Whbt + (size_t)gcol * H_))[seg];
    ((uint4*)(WhS + (size_t)row * WHS_STRIDE))[seg] = w;
  }
  // h(0) = 0, cooperatively across the whole grid
  for (int i = wg * 256 + tid; i < B_ * H_; i += NWG * 256) h0[i] = (bf16_t)0.0f;

  unsigned gen = 1;
  grid_barrier(cnt, gen * NWG);

  v8f cst = {};
  float hv[8];
#pragma unroll
  for (int r = 0; r < 8; ++r) hv[r] = 0.0f;

  for (int t = 0; t < T_; ++t) {
    const bf16_t* hcur  = (t & 1) ? h1 : h0;
    bf16_t*       hnext = (t & 1) ? h0 : h1;

    // seed accumulators with x_proj[t] tile (bias already folded in)
    v8f acc[4];
    const size_t tbase = (size_t)t * B_ * G4H;
#pragma unroll
    for (int q = 0; q < 4; ++q) {
      const size_t base = tbase + (size_t)q * 1024 + hc0 + ln;
#pragma unroll
      for (int r = 0; r < 8; ++r)
        acc[q][r] = (float)xproj[base + (size_t)(rowg0 + r) * G4H];
    }
    if (t + 1 < T_) {  // pull next step's x_proj tile toward the caches
      __builtin_prefetch(xproj + tbase + (size_t)B_ * G4H +
                             (size_t)rowg0 * G4H + hc0 + ln, 0, 1);
    }

    auto stageH = [&](int buf, int kb) {
#pragma unroll
      for (int it = 0; it < 2; ++it) {
        int e = tid + it * 256;        // 0..511
        int row = e >> 2;
        int seg = e & 3;
        uint4 u = *(const uint4*)(hcur + (size_t)row * H_ + kb + seg * 8);
        *(uint4*)&hS[buf][row][seg * 8] = u;
      }
    };

    stageH(0, 0);
    for (int ki = 0; ki < H_ / 32; ++ki) {
      __syncthreads();
      if (ki + 1 < H_ / 32) stageH((ki + 1) & 1, (ki + 1) * 32);
      const int buf = ki & 1;
      const int kb  = ki * 32;

      FragU fa;
      fa.q[0] = *(const uint4*)&hS[buf][rw0 + ln][hi * 8];
      fa.q[1] = *(const uint4*)&hS[buf][rw0 + ln][16 + hi * 8];
      FragU fb[4];
#pragma unroll
      for (int q = 0; q < 4; ++q) {
        const bf16_t* wrow = WhS + (size_t)(q * 16 + ln) * WHS_STRIDE + kb + hi * 8;
        fb[q].q[0] = *(const uint4*)wrow;
        fb[q].q[1] = *(const uint4*)(wrow + 16);
      }
#pragma unroll
      for (int q = 0; q < 4; ++q) {
        acc[q] = __builtin_amdgcn_wmma_f32_16x16x32_bf16(
            false, fa.v, false, fb[q].v, (short)0, acc[q], false, false);
      }
    }

    // gates -> state update, all in registers (accumulator layout)
#pragma unroll
    for (int r = 0; r < 8; ++r) {
      float ig = sigmoidf_(acc[0][r]);
      float fg = sigmoidf_(acc[1][r]);
      float gg = tanhf_(acc[2][r]);
      float og = sigmoidf_(acc[3][r]);
      float cv = fg * cst[r] + ig * gg;
      cst[r] = cv;
      hv[r]  = og * tanhf_(cv);
      hnext[(size_t)(rowg0 + r) * H_ + hc0 + ln] = (bf16_t)hv[r];
    }
    ++gen;
    grid_barrier(cnt, gen * NWG);
  }

  // final h_T, c_T in fp32
#pragma unroll
  for (int r = 0; r < 8; ++r) {
    out_h[(size_t)(rowg0 + r) * H_ + hc0 + ln] = hv[r];
    out_c[(size_t)(rowg0 + r) * H_ + hc0 + ln] = cst[r];
  }
}

// ---------------------------------------------------------------------------
extern "C" void kernel_launch(void* const* d_in, const int* in_sizes, int n_in,
                              void* d_out, int out_size, void* d_ws, size_t ws_size,
                              hipStream_t stream) {
  const float* x   = (const float*)d_in[0];
  const float* Wii = (const float*)d_in[1];
  const float* Wif = (const float*)d_in[2];
  const float* Wig = (const float*)d_in[3];
  const float* Wio = (const float*)d_in[4];
  const float* Whi = (const float*)d_in[5];
  const float* Whf = (const float*)d_in[6];
  const float* Whg = (const float*)d_in[7];
  const float* Who = (const float*)d_in[8];
  const float* b_i = (const float*)d_in[9];
  const float* b_f = (const float*)d_in[10];
  const float* b_g = (const float*)d_in[11];
  const float* b_o = (const float*)d_in[12];

  char* ws = (char*)d_ws;
  const size_t MB = 1024 * 1024;
  bf16_t* Wibt  = (bf16_t*)ws;                       // [4H,D]  8 MB
  bf16_t* Whbt  = (bf16_t*)(ws + 8 * MB);            // [4H,H]  8 MB
  float*  bias4 = (float*)(ws + 16 * MB);            // [4H]    16 KB
  bf16_t* xb    = (bf16_t*)(ws + 32 * MB);           // [T*B,D] 256 MB
  bf16_t* xproj = (bf16_t*)(ws + 288 * MB);          // [T*B,4H] 1 GB
  size_t xprojBytes = (size_t)T_ * B_ * G4H * sizeof(bf16_t);
  bf16_t* h0 = (bf16_t*)(ws + 288 * MB + xprojBytes);
  bf16_t* h1 = h0 + (size_t)B_ * H_;
  unsigned* cnt = (unsigned*)(h1 + (size_t)B_ * H_);

  hipMemsetAsync(cnt, 0, 64, stream);  // reset grid barrier (capture-legal)

  pack_x  <<<dim3(65536), dim3(256), 0, stream>>>(x, xb);
  pack_w4t<<<dim3(16384), dim3(256), 0, stream>>>(Wii, Wif, Wig, Wio, Wibt);
  pack_w4t<<<dim3(16384), dim3(256), 0, stream>>>(Whi, Whf, Whg, Who, Whbt);
  pack_bias<<<dim3(16),   dim3(256), 0, stream>>>(b_i, b_f, b_g, b_o, bias4);

  xproj_gemm<<<dim3(32, 1024), dim3(256), 0, stream>>>(xb, Wibt, bias4, xproj);

  size_t smem2 = (size_t)64 * WHS_STRIDE * sizeof(bf16_t) +
                 (size_t)2 * 128 * 40 * sizeof(bf16_t);  // ~149 KB
  lstm_recur<<<dim3(NWG), dim3(256), smem2, stream>>>(
      Whbt, xproj, h0, h1, cnt,
      (float*)d_out, (float*)d_out + (size_t)B_ * H_);
}
